// QuantQwen2AudioEncoder_3564822856087
// MI455X (gfx1250) — 16-bit and 8-bit matrix operand order repaired and checked by reading, compiled, not run on hardware
//
#include <hip/hip_runtime.h>
#include <cstdint>
#include <cstddef>

#define S_TOK   1500
#define S_PAD   1504
#define D_MODEL 1280
#define NHEAD   20
#define DHEAD   64
#define FFDIM   5120
#define NLAYER  4
#define TCONV   3000
#define NMEL    128

#define BK 64
#define BM 32
#define BN 256

typedef __attribute__((ext_vector_type(8)))  int          i32x8;
typedef __attribute__((ext_vector_type(4)))  int          i32x4;
typedef __attribute__((ext_vector_type(4)))  unsigned int u32x4;
typedef __attribute__((ext_vector_type(8)))  float        f32x8;
typedef __attribute__((ext_vector_type(16))) _Float16     f16x16;

#if __has_builtin(__builtin_amdgcn_tensor_load_to_lds) && \
    __has_builtin(__builtin_amdgcn_s_wait_tensorcnt)
#define HAVE_TDM 1
#else
#define HAVE_TDM 0
#endif

__device__ __forceinline__ float gelu_tanh(float x) {
  float x3 = x * x * x;
  return 0.5f * x * (1.0f + tanhf(0.79788456080286535588f * (x + 0.044715f * x3)));
}

__global__ __launch_bounds__(256)
void conv1_k(const float* __restrict__ in, const float* __restrict__ w,
             const float* __restrict__ b, float* __restrict__ out) {
  int idx = blockIdx.x * 256 + threadIdx.x;
  int co = idx / TCONV, t = idx % TCONV;
  float acc = b[co];
  for (int ci = 0; ci < NMEL; ++ci) {
    const float* ip = in + (size_t)ci * TCONV;
    const float* wp = w + ((size_t)co * NMEL + ci) * 3;
    if (t > 0)         acc += ip[t - 1] * wp[0];
    acc += ip[t] * wp[1];
    if (t < TCONV - 1) acc += ip[t + 1] * wp[2];
  }
  out[idx] = gelu_tanh(acc);
}

__global__ __launch_bounds__(256)
void conv2_k(const float* __restrict__ c1, const float* __restrict__ w,
             const float* __restrict__ b, const float* __restrict__ pos,
             float* __restrict__ h) {
  int idx = blockIdx.x * 256 + threadIdx.x;
  int t = idx / D_MODEL, co = idx % D_MODEL;
  if (t >= S_TOK) { h[idx] = 0.f; return; }
  int tc = 2 * t;
  float acc = b[co];
  for (int ci = 0; ci < D_MODEL; ++ci) {
    const float* ip = c1 + (size_t)ci * TCONV + tc;
    const float* wp = w + ((size_t)co * D_MODEL + ci) * 3;
    if (tc > 0) acc += ip[-1] * wp[0];
    acc += ip[0] * wp[1];
    acc += ip[1] * wp[2];
  }
  h[idx] = gelu_tanh(acc) + pos[idx];
}

__global__ __launch_bounds__(256)
void wquant_k(const float* __restrict__ w, int8_t* __restrict__ qw,
              float* __restrict__ wsc, int K) {
  __shared__ float red[256];
  const int r = blockIdx.x, tid = threadIdx.x;
  const float* row = w + (size_t)r * K;
  float mx = 0.f;
  for (int j = tid; j < K; j += 256) mx = fmaxf(mx, fabsf(row[j]));
  red[tid] = mx; __syncthreads();
  for (int s = 128; s > 0; s >>= 1) { if (tid < s) red[tid] = fmaxf(red[tid], red[tid + s]); __syncthreads(); }
  const float sc = fmaxf(red[0] / 127.f, 1e-8f);
  if (tid == 0) wsc[r] = sc;
  const float is = 1.f / sc;
  for (int j = tid; j < K; j += 256) {
    float v = rintf(row[j] * is);
    v = fminf(fmaxf(v, -128.f), 127.f);
    qw[(size_t)r * K + j] = (int8_t)v;
  }
}

__global__ __launch_bounds__(256)
void rmsnorm_quant_k(const float* __restrict__ x, const float* __restrict__ w,
                     const float* __restrict__ b, int8_t* __restrict__ q,
                     float* __restrict__ sx) {
  __shared__ float red[256];
  const int r = blockIdx.x, tid = threadIdx.x;
  const int K = D_MODEL;
  if (r >= S_TOK) {
    for (int j = tid; j < K; j += 256) q[(size_t)r * K + j] = 0;
    if (tid == 0) sx[r] = 1.f;
    return;
  }
  const float* row = x + (size_t)r * K;
  float ss = 0.f;
  for (int j = tid; j < K; j += 256) { float v = row[j]; ss += v * v; }
  red[tid] = ss; __syncthreads();
  for (int s = 128; s > 0; s >>= 1) { if (tid < s) red[tid] += red[tid + s]; __syncthreads(); }
  const float inv = rsqrtf(red[0] / K + 1e-6f);
  __syncthreads();
  float mx = 0.f;
  for (int j = tid; j < K; j += 256) {
    float v = row[j] * inv * w[j] + b[j];
    mx = fmaxf(mx, fabsf(v));
  }
  red[tid] = mx; __syncthreads();
  for (int s = 128; s > 0; s >>= 1) { if (tid < s) red[tid] = fmaxf(red[tid], red[tid + s]); __syncthreads(); }
  const float sc = fmaxf(red[0] / 127.f, 1e-8f);
  if (tid == 0) sx[r] = sc;
  const float is = 1.f / sc;
  for (int j = tid; j < K; j += 256) {
    float v = rintf((row[j] * inv * w[j] + b[j]) * is);
    v = fminf(fmaxf(v, -128.f), 127.f);
    q[(size_t)r * K + j] = (int8_t)v;
  }
}

__global__ __launch_bounds__(256)
void act_quant_k(const float* __restrict__ x, int8_t* __restrict__ q,
                 float* __restrict__ sx, int K, int do_gelu) {
  __shared__ float red[256];
  const int r = blockIdx.x, tid = threadIdx.x;
  if (r >= S_TOK) {
    for (int j = tid; j < K; j += 256) q[(size_t)r * K + j] = 0;
    if (tid == 0) sx[r] = 1.f;
    return;
  }
  const float* row = x + (size_t)r * K;
  float mx = 0.f;
  for (int j = tid; j < K; j += 256) {
    float v = row[j];
    if (do_gelu) v = gelu_tanh(v);
    mx = fmaxf(mx, fabsf(v));
  }
  red[tid] = mx; __syncthreads();
  for (int s = 128; s > 0; s >>= 1) { if (tid < s) red[tid] = fmaxf(red[tid], red[tid + s]); __syncthreads(); }
  const float sc = fmaxf(red[0] / 127.f, 1e-8f);
  if (tid == 0) sx[r] = sc;
  const float is = 1.f / sc;
  for (int j = tid; j < K; j += 256) {
    float v = row[j];
    if (do_gelu) v = gelu_tanh(v);
    v = rintf(v * is);
    v = fminf(fmaxf(v, -128.f), 127.f);
    q[(size_t)r * K + j] = (int8_t)v;
  }
}

__global__ __launch_bounds__(256)
void gemm_w8a8(const int8_t* __restrict__ qx, const float* __restrict__ sx,
               const int8_t* __restrict__ qw, const float* __restrict__ wsc,
               const float* __restrict__ bias, const float* __restrict__ res,
               float* __restrict__ out, int M, int N, int K) {
  __shared__ int8_t ldsA[2][BM * BK];
  const int lane = threadIdx.x & 31;
  const int wave = threadIdx.x >> 5;
  const int ntn = N / BN;
  const int m0 = (blockIdx.x / ntn) * BM;
  const int n0 = (blockIdx.x % ntn) * BN + wave * 32;
  const int mrow = lane & 15, half = lane >> 4;
  const int nk = K / BK;

#if HAVE_TDM
  auto tdm_issue = [&](int ks, int buf) {
    uint64_t ga = (uint64_t)(uintptr_t)(qx + (size_t)m0 * K + (size_t)ks * BK);
    uint32_t lds_off = (uint32_t)(uintptr_t)&ldsA[buf][0];
    u32x4 g0;
    g0[0] = 1u;
    g0[1] = lds_off;
    g0[2] = (uint32_t)ga;
    g0[3] = (uint32_t)((ga >> 32) & 0x1FFFFFFu) | (2u << 30);
    i32x8 g1;
    g1[0] = 0;
    g1[1] = (int)(((uint32_t)K & 0xFFFFu) << 16);
    g1[2] = (int)((((uint32_t)K >> 16) & 0xFFFFu) |
                  (((uint32_t)M & 0xFFFFu) << 16));
    g1[3] = (int)((((uint32_t)M >> 16) & 0xFFFFu) |
                  ((uint32_t)BK << 16));
    g1[4] = (int)(uint32_t)BM;
    g1[5] = (int)(uint32_t)K;
    g1[6] = 0;
    g1[7] = 0;
    i32x4 z4 = {};
#if __clang_major__ >= 23
    i32x8 z8 = {};
    __builtin_amdgcn_tensor_load_to_lds(g0, g1, z4, z4, z8, 0);
#else
    __builtin_amdgcn_tensor_load_to_lds(g0, g1, z4, z4, 0);
#endif
  };
  if (wave == 0) tdm_issue(0, 0);
#endif

  i32x8 acc[2][2] = {};
  for (int ks = 0; ks < nk; ++ks) {
#if HAVE_TDM
    const int cur = ks & 1;
    if (wave == 0) {
      if (ks + 1 < nk) {
        tdm_issue(ks + 1, cur ^ 1);
        __builtin_amdgcn_s_wait_tensorcnt(1);
      } else {
        __builtin_amdgcn_s_wait_tensorcnt(0);
      }
    }
    __syncthreads();
#else
    const int cur = 0;
    __syncthreads();
    {
      int o = threadIdx.x * 8;
      int rrow = o >> 6, col = o & 63;
      const int8_t* src = qx + (size_t)(m0 + rrow) * K + (size_t)ks * BK + col;
      int* dst = (int*)&ldsA[0][o];
      dst[0] = ((const int*)src)[0];
      dst[1] = ((const int*)src)[1];
    }
    __syncthreads();
#endif

    i32x8 a0, a1, b0, b1;
    const int8_t* pa0 = &ldsA[cur][0] + mrow * BK + 16 * half;
    const int8_t* pa1 = pa0 + 16 * BK;
#pragma unroll
    for (int pp = 0; pp < 4; ++pp) {
      const int* q0 = (const int*)(pa0 + 8 * pp + 16 * (pp >> 1));
      const int* q1 = (const int*)(pa1 + 8 * pp + 16 * (pp >> 1));
      a0[2 * pp] = q0[0]; a0[2 * pp + 1] = q0[1];
      a1[2 * pp] = q1[0]; a1[2 * pp + 1] = q1[1];
    }
    const int8_t* pb0 = qw + (size_t)(n0 + mrow) * K + (size_t)ks * BK + 16 * half;
    const int8_t* pb1 = pb0 + (size_t)16 * K;
#pragma unroll
    for (int i = 0; i < 4; ++i) {
      b0[i]     = ((const int*)pb0)[i];
      b0[4 + i] = ((const int*)(pb0 + 32))[i];
      b1[i]     = ((const int*)pb1)[i];
      b1[4 + i] = ((const int*)(pb1 + 32))[i];
    }
    acc[0][0] = __builtin_amdgcn_wmma_i32_16x16x64_iu8(true, a0, true, b0, acc[0][0], false, false);
    acc[0][1] = __builtin_amdgcn_wmma_i32_16x16x64_iu8(true, a0, true, b1, acc[0][1], false, false);
    acc[1][0] = __builtin_amdgcn_wmma_i32_16x16x64_iu8(true, a1, true, b0, acc[1][0], false, false);
    acc[1][1] = __builtin_amdgcn_wmma_i32_16x16x64_iu8(true, a1, true, b1, acc[1][1], false, false);

#if HAVE_TDM
    __syncthreads();
#endif
  }

#pragma unroll
  for (int i = 0; i < 2; ++i)
#pragma unroll
    for (int j = 0; j < 2; ++j) {
      const int nn = n0 + 16 * j + mrow;
      const float wv = wsc[nn];
      const float bb = bias ? bias[nn] : 0.f;
#pragma unroll
      for (int r = 0; r < 8; ++r) {
        const int mm = m0 + 16 * i + r + 8 * half;
        float v = (float)acc[i][j][r] * sx[mm] * wv + bb;
        if (res) v += res[(size_t)mm * N + nn];
        out[(size_t)mm * N + nn] = v;
      }
    }
}

__global__ __launch_bounds__(256)
void split_qkv_k(const float* __restrict__ qkvo, _Float16* __restrict__ Qh,
                 _Float16* __restrict__ Kh, _Float16* __restrict__ Vt) {
  int idx = blockIdx.x * 256 + threadIdx.x;
  int t = idx / D_MODEL, c = idx % D_MODEL;
  int hh = c / DHEAD, d = c % DHEAD;
  float qv, kv, vv;
  if (t < S_TOK) {
    const float* p = qkvo + (size_t)t * 3 * D_MODEL;
    qv = p[c]; kv = p[D_MODEL + c]; vv = p[2 * D_MODEL + c];
  } else { qv = kv = vv = 0.f; }
  Qh[((size_t)hh * S_PAD + t) * DHEAD + d] = (_Float16)qv;
  Kh[((size_t)hh * S_PAD + t) * DHEAD + d] = (_Float16)kv;
  Vt[((size_t)hh * DHEAD + d) * S_PAD + t] = (_Float16)vv;
}

__global__ __launch_bounds__(256)
void attn_scores_k(const _Float16* __restrict__ Qh, const _Float16* __restrict__ Kh,
                   float* __restrict__ sc, float scale) {
  const int lane = threadIdx.x & 31;
  const int tile = blockIdx.x * 8 + (threadIdx.x >> 5);
  const int nt = S_PAD / 16;
  if (tile >= nt * nt) return;
  const int m0 = (tile / nt) * 16, n0 = (tile % nt) * 16;
  const int mrow = lane & 15, half = lane >> 4;
  f32x8 acc = {};
#pragma unroll
  for (int k0 = 0; k0 < DHEAD; k0 += 32) {
    union { i32x8 i; f16x16 h; } ua, ub;
    const int* qa  = (const int*)(Qh + (size_t)(m0 + mrow) * DHEAD + k0 + 16 * half);
    const int* qa2 = qa + 4;
#pragma unroll
    for (int i = 0; i < 4; ++i) { ua.i[i] = qa[i]; ua.i[4 + i] = qa2[i]; }
    const int* qb = (const int*)(Kh + (size_t)(n0 + mrow) * DHEAD + k0 + 16 * half);
#pragma unroll
    for (int i = 0; i < 8; ++i) ub.i[i] = qb[i];
    acc = __builtin_amdgcn_wmma_f32_16x16x32_f16(false, ua.h, false, ub.h, (short)0, acc, false, false);
  }
#pragma unroll
  for (int r = 0; r < 8; ++r) {
    const int mm = m0 + r + 8 * half, nn = n0 + mrow;
    sc[(size_t)mm * S_PAD + nn] = acc[r] * scale;
  }
}

__global__ __launch_bounds__(256)
void softmax_k(const float* __restrict__ sc, _Float16* __restrict__ at) {
  __shared__ float red[256];
  const int r = blockIdx.x, tid = threadIdx.x;
  if (r >= S_TOK) {
    for (int j = tid; j < S_PAD; j += 256) at[(size_t)r * S_PAD + j] = (_Float16)0.f;
    return;
  }
  const float* row = sc + (size_t)r * S_PAD;
  float m = -1e30f;
  for (int j = tid; j < S_TOK; j += 256) m = fmaxf(m, row[j]);
  red[tid] = m; __syncthreads();
  for (int s = 128; s > 0; s >>= 1) { if (tid < s) red[tid] = fmaxf(red[tid], red[tid + s]); __syncthreads(); }
  m = red[0]; __syncthreads();
  float sum = 0.f;
  for (int j = tid; j < S_TOK; j += 256) sum += __expf(row[j] - m);
  red[tid] = sum; __syncthreads();
  for (int s = 128; s > 0; s >>= 1) { if (tid < s) red[tid] += red[tid + s]; __syncthreads(); }
  const float inv = 1.f / red[0];
  for (int j = tid; j < S_PAD; j += 256)
    at[(size_t)r * S_PAD + j] = (j < S_TOK) ? (_Float16)(__expf(row[j] - m) * inv)
                                            : (_Float16)0.f;
}

__global__ __launch_bounds__(256)
void attn_av_k(const _Float16* __restrict__ at, const _Float16* __restrict__ Vt,
               float* __restrict__ ao, int head) {
  const int lane = threadIdx.x & 31;
  const int tile = blockIdx.x * 8 + (threadIdx.x >> 5);
  const int ntn = DHEAD / 16;
  if (tile >= (S_PAD / 16) * ntn) return;
  const int m0 = (tile / ntn) * 16, n0 = (tile % ntn) * 16;
  const int mrow = lane & 15, half = lane >> 4;
  f32x8 acc = {};
  for (int k0 = 0; k0 < S_PAD; k0 += 32) {
    union { i32x8 i; f16x16 h; } ua, ub;
    const int* qa  = (const int*)(at + (size_t)(m0 + mrow) * S_PAD + k0 + 16 * half);
    const int* qa2 = qa + 4;
#pragma unroll
    for (int i = 0; i < 4; ++i) { ua.i[i] = qa[i]; ua.i[4 + i] = qa2[i]; }
    const int* qb = (const int*)(Vt + (size_t)(n0 + mrow) * S_PAD + k0 + 16 * half);
#pragma unroll
    for (int i = 0; i < 8; ++i) ub.i[i] = qb[i];
    acc = __builtin_amdgcn_wmma_f32_16x16x32_f16(false, ua.h, false, ub.h, (short)0, acc, false, false);
  }
#pragma unroll
  for (int r = 0; r < 8; ++r) {
    const int mm = m0 + r + 8 * half;
    ao[(size_t)mm * D_MODEL + head * DHEAD + n0 + mrow] = acc[r];
  }
}

__global__ __launch_bounds__(256)
void final_k(const float* __restrict__ h, const float* __restrict__ lnw,
             const float* __restrict__ lnb, float* __restrict__ out) {
  __shared__ float red[256];
  const int r = blockIdx.x, tid = threadIdx.x;
  const float* r0 = h + (size_t)(2 * r) * D_MODEL;
  const float* r1 = r0 + D_MODEL;
  float t[5];
  float sum = 0.f;
#pragma unroll
  for (int i = 0; i < 5; ++i) {
    int j = tid + i * 256;
    float v = 0.5f * (r0[j] + r1[j]);
    t[i] = v; sum += v;
  }
  red[tid] = sum; __syncthreads();
  for (int s = 128; s > 0; s >>= 1) { if (tid < s) red[tid] += red[tid + s]; __syncthreads(); }
  const float mu = red[0] / D_MODEL; __syncthreads();
  float vs = 0.f;
#pragma unroll
  for (int i = 0; i < 5; ++i) { float d = t[i] - mu; vs += d * d; }
  red[tid] = vs; __syncthreads();
  for (int s = 128; s > 0; s >>= 1) { if (tid < s) red[tid] += red[tid + s]; __syncthreads(); }
  const float inv = rsqrtf(red[0] / D_MODEL + 1e-5f);
#pragma unroll
  for (int i = 0; i < 5; ++i) {
    int j = tid + i * 256;
    out[(size_t)r * D_MODEL + j] = (t[i] - mu) * inv * lnw[j] + lnb[j];
  }
}

extern "C" void kernel_launch(void* const* d_in, const int* in_sizes, int n_in,
                              void* d_out, int out_size, void* d_ws, size_t ws_size,
                              hipStream_t stream) {
  (void)in_sizes; (void)n_in; (void)out_size; (void)ws_size;
  const float* in_feat = (const float*)d_in[0];
  const float* c1w  = (const float*)d_in[1];
  const float* c1b  = (const float*)d_in[2];
  const float* c2w  = (const float*)d_in[3];
  const float* c2b  = (const float*)d_in[4];
  const float* pos  = (const float*)d_in[5];
  const float* ln1w = (const float*)d_in[6];
  const float* ln1b = (const float*)d_in[7];
  const float* qkvw = (const float*)d_in[8];
  const float* qkvb = (const float*)d_in[9];
  const float* outw = (const float*)d_in[10];
  const float* outb = (const float*)d_in[11];
  const float* ln2w = (const float*)d_in[12];
  const float* ln2b = (const float*)d_in[13];
  const float* f1w  = (const float*)d_in[14];
  const float* f1b  = (const float*)d_in[15];
  const float* f2w  = (const float*)d_in[16];
  const float* f2b  = (const float*)d_in[17];
  const float* lnfw = (const float*)d_in[18];
  const float* lnfb = (const float*)d_in[19];

  char* wp = (char*)d_ws;
  auto alloc = [&](size_t bytes) -> void* {
    void* r = (void*)wp;
    wp += (bytes + 255) & ~(size_t)255;
    return r;
  };
  float*    c1   = (float*)alloc((size_t)D_MODEL * TCONV * 4);
  float*    hA   = (float*)alloc((size_t)S_PAD * D_MODEL * 4);
  float*    hB   = (float*)alloc((size_t)S_PAD * D_MODEL * 4);
  int8_t*   qx   = (int8_t*)alloc((size_t)S_PAD * FFDIM);
  float*    sx   = (float*)alloc((size_t)S_PAD * 4);
  int8_t*   qwq  = (int8_t*)alloc((size_t)3 * D_MODEL * D_MODEL);
  float*    wsq  = (float*)alloc((size_t)3 * D_MODEL * 4);
  int8_t*   qwo  = (int8_t*)alloc((size_t)D_MODEL * D_MODEL);
  float*    wso  = (float*)alloc((size_t)D_MODEL * 4);
  int8_t*   qw1  = (int8_t*)alloc((size_t)FFDIM * D_MODEL);
  float*    ws1  = (float*)alloc((size_t)FFDIM * 4);
  int8_t*   qw2  = (int8_t*)alloc((size_t)D_MODEL * FFDIM);
  float*    ws2  = (float*)alloc((size_t)D_MODEL * 4);
  float*    qkvo = (float*)alloc((size_t)S_PAD * 3 * D_MODEL * 4);
  _Float16* Qh   = (_Float16*)alloc((size_t)NHEAD * S_PAD * DHEAD * 2);
  _Float16* Kh   = (_Float16*)alloc((size_t)NHEAD * S_PAD * DHEAD * 2);
  _Float16* Vt   = (_Float16*)alloc((size_t)NHEAD * DHEAD * S_PAD * 2);
  float*    sc   = (float*)alloc((size_t)S_PAD * S_PAD * 4);
  _Float16* at   = (_Float16*)alloc((size_t)S_PAD * S_PAD * 2);
  float*    ao   = (float*)alloc((size_t)S_PAD * D_MODEL * 4);
  float*    f1o  = (float*)alloc((size_t)S_PAD * FFDIM * 4);

  auto gb = [](int tiles) { return (tiles + 7) / 8; };
  const float scale = 0.125f;

  conv1_k<<<(D_MODEL * TCONV) / 256, 256, 0, stream>>>(in_feat, c1w, c1b, c1);
  conv2_k<<<(S_PAD * D_MODEL) / 256, 256, 0, stream>>>(c1, c2w, c2b, pos, hA);

  for (int l = 0; l < NLAYER; ++l) {
    wquant_k<<<3 * D_MODEL, 256, 0, stream>>>(qkvw + (size_t)l * 3 * D_MODEL * D_MODEL, qwq, wsq, D_MODEL);
    wquant_k<<<D_MODEL,     256, 0, stream>>>(outw + (size_t)l * D_MODEL * D_MODEL,     qwo, wso, D_MODEL);
    wquant_k<<<FFDIM,       256, 0, stream>>>(f1w  + (size_t)l * FFDIM * D_MODEL,       qw1, ws1, D_MODEL);
    wquant_k<<<D_MODEL,     256, 0, stream>>>(f2w  + (size_t)l * D_MODEL * FFDIM,       qw2, ws2, FFDIM);

    rmsnorm_quant_k<<<S_PAD, 256, 0, stream>>>(hA, ln1w + (size_t)l * D_MODEL,
                                               ln1b + (size_t)l * D_MODEL, qx, sx);
    gemm_w8a8<<<(S_PAD / BM) * (3 * D_MODEL / BN), 256, 0, stream>>>(
        qx, sx, qwq, wsq, qkvb + (size_t)l * 3 * D_MODEL, nullptr, qkvo,
        S_PAD, 3 * D_MODEL, D_MODEL);
    split_qkv_k<<<(S_PAD * D_MODEL) / 256, 256, 0, stream>>>(qkvo, Qh, Kh, Vt);
    for (int hh = 0; hh < NHEAD; ++hh) {
      attn_scores_k<<<gb((S_PAD / 16) * (S_PAD / 16)), 256, 0, stream>>>(
          Qh + (size_t)hh * S_PAD * DHEAD, Kh + (size_t)hh * S_PAD * DHEAD, sc, scale);
      softmax_k<<<S_PAD, 256, 0, stream>>>(sc, at);
      attn_av_k<<<gb((S_PAD / 16) * (DHEAD / 16)), 256, 0, stream>>>(
          at, Vt + (size_t)hh * DHEAD * S_PAD, ao, hh);
    }
    act_quant_k<<<S_PAD, 256, 0, stream>>>(ao, qx, sx, D_MODEL, 0);
    gemm_w8a8<<<(S_PAD / BM) * (D_MODEL / BN), 256, 0, stream>>>(
        qx, sx, qwo, wso, outb + (size_t)l * D_MODEL, hA, hB,
        S_PAD, D_MODEL, D_MODEL);

    rmsnorm_quant_k<<<S_PAD, 256, 0, stream>>>(hB, ln2w + (size_t)l * D_MODEL,
                                               ln2b + (size_t)l * D_MODEL, qx, sx);
    gemm_w8a8<<<(S_PAD / BM) * (FFDIM / BN), 256, 0, stream>>>(
        qx, sx, qw1, ws1, f1b + (size_t)l * FFDIM, nullptr, f1o,
        S_PAD, FFDIM, D_MODEL);
    act_quant_k<<<S_PAD, 256, 0, stream>>>(f1o, qx, sx, FFDIM, 1);
    gemm_w8a8<<<(S_PAD / BM) * (D_MODEL / BN), 256, 0, stream>>>(
        qx, sx, qw2, ws2, f2b + (size_t)l * D_MODEL, hB, hA,
        S_PAD, D_MODEL, FFDIM);
  }

  final_k<<<S_TOK / 2, 256, 0, stream>>>(hA, lnfw, lnfb, (float*)d_out);
}
